// GEFormerDTA_89704686944431
// MI455X (gfx1250) — compile-verified, matrix operand order checked by reading, not run on hardware
//
#include <hip/hip_runtime.h>
#include <hip/hip_bf16.h>
#include <math.h>

// ---------------------------------------------------------------------------
// Problem constants (from the reference)
// ---------------------------------------------------------------------------
#define B_     4
#define L_     2048
#define C_     256
#define H_     8
#define E_     32
#define NL_    3
#define INTER_ 16
#define NTOP_  40                  // 5 * ceil(ln(2048)) = 5 * 8
#define MROWS  (B_ * L_)           // 8192 token rows
#define SCALE_ 0.17677669529663687f // 1/sqrt(32)

// ---------------------------------------------------------------------------
// CDNA5 WMMA types
// ---------------------------------------------------------------------------
typedef __attribute__((ext_vector_type(16))) __bf16 v16bf;
typedef __attribute__((ext_vector_type(8)))  float  v8f;

union BF16Frag { unsigned short u[16]; v16bf v; };

__device__ __forceinline__ unsigned short f2bf(float f) {
  union { float f; unsigned u; } x; x.f = f;
  unsigned r = x.u + 0x7FFFu + ((x.u >> 16) & 1u); // round-to-nearest-even
  return (unsigned short)(r >> 16);
}

// CDNA5 async global->LDS copy (ASYNCcnt-tracked, bypasses VGPRs).
// Two b128 transfers from one 64-bit base; second uses IOFFSET immediate.
__device__ __forceinline__ void async_load_b128x2(void* lds0, void* lds1,
                                                  const void* gaddr) {
  unsigned l0 = (unsigned)(__SIZE_TYPE__)lds0;   // addr[31:0] = LDS offset
  unsigned l1 = (unsigned)(__SIZE_TYPE__)lds1;
  unsigned long long ga = (unsigned long long)(__SIZE_TYPE__)gaddr;
  asm volatile("global_load_async_to_lds_b128 %0, %2, off\n\t"
               "global_load_async_to_lds_b128 %1, %2, off offset:16"
               :: "v"(l0), "v"(l1), "v"(ga) : "memory");
}
__device__ __forceinline__ void wait_async0() {
  asm volatile("s_wait_asynccnt 0" ::: "memory");
}
__device__ __forceinline__ void wait_async4() {   // older 4-load group landed
  asm volatile("s_wait_asynccnt 4" ::: "memory");
}

// ---------------------------------------------------------------------------
// Elementwise helpers
// ---------------------------------------------------------------------------
__global__ __launch_bounds__(256) void copy_f32_kernel(const float* __restrict__ in,
                                                       float* __restrict__ out, int n) {
  int i = blockIdx.x * 256 + threadIdx.x;
  if (i < n) out[i] = in[i];
}

__global__ __launch_bounds__(256) void cvt_bf16_kernel(const float* __restrict__ in,
                                                       unsigned short* __restrict__ out, int n) {
  int i = blockIdx.x * 256 + threadIdx.x;
  if (i < n) out[i] = f2bf(in[i]);
}

// transpose+convert: in is KxN (row-major), out is NxK (row-major)
__global__ __launch_bounds__(256) void cvtT_bf16_kernel(const float* __restrict__ in,
                                                        unsigned short* __restrict__ out,
                                                        int K, int N) {
  int i = blockIdx.x * 256 + threadIdx.x;
  if (i >= K * N) return;
  int k = i % K, n = i / K;          // i indexes output [n][k]
  out[i] = f2bf(in[(size_t)k * N + n]);
}

// im2col for the 3-tap circular conv: out[m][t*C + c] = x[b, (l+t-1) mod L, c]
__global__ __launch_bounds__(256) void im2col_kernel(const float* __restrict__ x,
                                                     unsigned short* __restrict__ out) {
  int n = blockIdx.x * 256 + threadIdx.x;
  if (n >= MROWS * 3 * C_) return;
  int c = n % C_;
  int t = (n / C_) % 3;
  int m = n / (3 * C_);
  int b = m >> 11, l = m & (L_ - 1);
  int ls = l + t - 1;
  ls = (ls < 0) ? (L_ - 1) : ((ls >= L_) ? 0 : ls);
  out[n] = f2bf(x[((size_t)(b * L_ + ls)) * C_ + c]);
}

// conv weight (O,I,3) -> bf16 Wt[o][t*C+ci]  (transposed column layout for GEMM B)
__global__ __launch_bounds__(256) void convw_kernel(const float* __restrict__ w,
                                                    unsigned short* __restrict__ out) {
  int n = blockIdx.x * 256 + threadIdx.x;
  if (n >= 3 * C_ * C_) return;
  int kidx = n % (3 * C_);           // t*256 + ci
  int o    = n / (3 * C_);
  int t    = kidx >> 8;
  int ci   = kidx & 255;
  out[n] = f2bf(w[((size_t)o * C_ + ci) * 3 + t]);
}

// ---------------------------------------------------------------------------
// WMMA GEMM: Y[M x N] = A(bf16)[M x K] @ B, where B is given TRANSPOSED:
// Bt(bf16)[N x K].  (+bias)(+resid)(relu)
// Block: 256 thr = 8 waves; block tile 128x128; wave tile 32x64 (2x4 WMMA)
// Double-buffered LDS pipeline: tile t+1 staged via
// global_load_async_to_lds_b128 while tile t is computed; s_wait_asynccnt 4
// retires only the older 4-load group (ASYNCcnt completes in order).
// qkv_mode scatters output into (B,H,L,E) layout.
// ---------------------------------------------------------------------------
#define TM  128
#define TN  128
#define TK  32
#define LDT 40   // padded row stride (ushort) -> 80B rows, 16B aligned

__global__ __launch_bounds__(256)
void wmma_gemm_kernel(const unsigned short* __restrict__ A,
                      const unsigned short* __restrict__ Bt,
                      const float* __restrict__ bias,
                      const float* __restrict__ resid,
                      float* __restrict__ Y,
                      int M, int K, int N,
                      int qkv_mode, int relu_mode)
{
  __shared__ unsigned short As[2][TM][LDT];   // 2 x 128 (m) x 32 (k) bf16
  __shared__ unsigned short Bs[2][TN][LDT];   // 2 x 128 (n) x 32 (k) bf16

  const int tid  = threadIdx.x;
  const int lane = tid & 31;
  const int wave = tid >> 5;
  const int wm   = wave & 3;       // 4 waves along M (32 rows each)
  const int wn   = wave >> 2;      // 2 waves along N (64 cols each)
  const int m0   = blockIdx.y * TM;
  const int n0   = blockIdx.x * TN;
  const int lrow = lane & 15;
  const int khi  = (lane >> 4) << 3;     // +8 K offset for upper half-wave

  const int row = tid >> 1, seg = tid & 1;   // 128 rows x 2 x 16 bf16 (A and Bt)

  const unsigned short* gaBase = A  + (size_t)(m0 + row) * K + seg * 16;
  const unsigned short* gbBase = Bt + (size_t)(n0 + row) * K + seg * 16;

  v8f acc[2][4] = {};

  const int nsteps = K / TK;

  // prologue: stage tile 0 into buffer 0
  async_load_b128x2(&As[0][row][seg * 16], &As[0][row][seg * 16 + 8], gaBase);
  async_load_b128x2(&Bs[0][row][seg * 16], &Bs[0][row][seg * 16 + 8], gbBase);

  for (int t = 0; t < nsteps; ++t) {
    const int cur = t & 1;
    if (t + 1 < nsteps) {
      const int nxt = cur ^ 1;
      const int kn  = (t + 1) * TK;
      async_load_b128x2(&As[nxt][row][seg * 16], &As[nxt][row][seg * 16 + 8],
                        gaBase + kn);
      async_load_b128x2(&Bs[nxt][row][seg * 16], &Bs[nxt][row][seg * 16 + 8],
                        gbBase + kn);
      if (t + 2 < nsteps)  // hint tile t+2 A slab (global_prefetch_b8)
        __builtin_prefetch(gaBase + (t + 2) * TK, 0, 1);
      wait_async4();   // the 4 loads of tile t have landed
    } else {
      wait_async0();
    }
    __syncthreads();   // all waves' LDS writes for tile t visible

    // ---- build fragments: K(e) = e + (e>=8 ? 8 : 0) + khi; two 16B chunks ----
    BF16Frag af[2], bf[4];
#pragma unroll
    for (int i = 0; i < 2; ++i) {
      const int r0 = wm * 32 + i * 16 + lrow;
#pragma unroll
      for (int e = 0; e < 16; ++e) {
        const int kk = e + ((e >> 3) << 3) + khi;
        af[i].u[e] = As[cur][r0][kk];
      }
    }
#pragma unroll
    for (int j = 0; j < 4; ++j) {
      const int c0 = wn * 64 + j * 16 + lrow;
#pragma unroll
      for (int e = 0; e < 16; ++e) {
        const int kk = e + ((e >> 3) << 3) + khi;
        bf[j].u[e] = Bs[cur][c0][kk];
      }
    }

#pragma unroll
    for (int i = 0; i < 2; ++i)
#pragma unroll
      for (int j = 0; j < 4; ++j)
        acc[i][j] = __builtin_amdgcn_wmma_f32_16x16x32_bf16(
            false, af[i].v, false, bf[j].v, (short)0, acc[i][j], false, false);

    __syncthreads();   // reads of buffer `cur` done before it is re-staged
  }

  // ---- epilogue: bias / residual / relu / layout, per C/D VGPR layout ----
  const int rhalf = (lane >> 4) << 3;   // rows 0..7 or 8..15
#pragma unroll
  for (int i = 0; i < 2; ++i)
#pragma unroll
    for (int j = 0; j < 4; ++j)
#pragma unroll
      for (int r = 0; r < 8; ++r) {
        int orow = m0 + wm * 32 + i * 16 + r + rhalf;
        int ocol = n0 + wn * 64 + j * 16 + lrow;
        float val = acc[i][j][r];
        if (bias)  val += bias[ocol];
        if (resid) val += resid[(size_t)orow * N + ocol];
        if (relu_mode) val = fmaxf(val, 0.f);
        if (qkv_mode) {
          int b = orow >> 11, l = orow & (L_ - 1);
          int h = ocol >> 5,  e = ocol & 31;
          Y[(((size_t)(b * H_ + h)) * L_ + l) * E_ + e] = val;
        } else {
          Y[(size_t)orow * N + ocol] = val;
        }
      }
}

// ---------------------------------------------------------------------------
// ProbSparse attention pieces
// ---------------------------------------------------------------------------
// M[b,h,l] = max_s(q.k_idx[s]) - (1/L) * sum_s(q.k_idx[s])
__global__ __launch_bounds__(256)
void sampled_m_kernel(const float* __restrict__ q, const float* __restrict__ k,
                      const int* __restrict__ idx, float* __restrict__ Mout)
{
  int t = blockIdx.x * 256 + threadIdx.x;
  if (t >= B_ * H_ * L_) return;
  int l  = t & (L_ - 1);
  int bh = t >> 11;
  const float* qp = q + (size_t)t * E_;
  const int*   ip = idx + (size_t)l * NTOP_;
  float mx = -3.4e38f, sm = 0.f;
  for (int s = 0; s < NTOP_; ++s) {
    int j = ip[s];
    const float* kp = k + ((size_t)bh * L_ + j) * E_;
    float d = 0.f;
#pragma unroll
    for (int e = 0; e < E_; ++e) d += qp[e] * kp[e];
    mx = fmaxf(mx, d);
    sm += d;
  }
  Mout[t] = mx - sm * (1.0f / (float)L_);
}

// iterative top-NTOP per (b,h); lowest index wins ties (matches lax.top_k)
__global__ __launch_bounds__(256)
void topk_kernel(const float* __restrict__ Mbuf, int* __restrict__ top)
{
  __shared__ float Ms[L_];
  __shared__ float rv[256];
  __shared__ int   ri[256];
  int bh = blockIdx.x, tid = threadIdx.x;
  for (int j = tid; j < L_; j += 256) Ms[j] = Mbuf[(size_t)bh * L_ + j];
  __syncthreads();
  for (int t = 0; t < NTOP_; ++t) {
    float bv = -3.4e38f; int bi = 0;
    for (int j = tid; j < L_; j += 256)
      if (Ms[j] > bv) { bv = Ms[j]; bi = j; }
    rv[tid] = bv; ri[tid] = bi;
    __syncthreads();
    for (int s = 128; s > 0; s >>= 1) {
      if (tid < s) {
        if (rv[tid + s] > rv[tid] ||
            (rv[tid + s] == rv[tid] && ri[tid + s] < ri[tid])) {
          rv[tid] = rv[tid + s]; ri[tid] = ri[tid + s];
        }
      }
      __syncthreads();
    }
    if (tid == 0) { top[bh * NTOP_ + t] = ri[0]; Ms[ri[0]] = -3.4e38f; }
    __syncthreads();
  }
}

// vmean[b,h,e] = mean_l v[b,h,l,e]
__global__ __launch_bounds__(256)
void vmean_kernel(const float* __restrict__ v, float* __restrict__ vm)
{
  __shared__ float acc[8][E_];
  int bh = blockIdx.x, tid = threadIdx.x;
  int e = tid & 31, chunk = tid >> 5;
  float s = 0.f;
  for (int l = chunk; l < L_; l += 8)
    s += v[((size_t)bh * L_ + l) * E_ + e];
  acc[chunk][e] = s;
  __syncthreads();
  if (tid < E_) {
    float t = 0.f;
    for (int c = 0; c < 8; ++c) t += acc[c][tid];
    vm[bh * E_ + tid] = t * (1.0f / (float)L_);
  }
}

// ctx(B,L,C) = broadcast vmean
__global__ __launch_bounds__(256)
void ctx_fill_kernel(const float* __restrict__ vm, float* __restrict__ ctx)
{
  int n = blockIdx.x * 256 + threadIdx.x;
  if (n >= MROWS * C_) return;
  int c = n & 255;
  int m = n >> 8;
  int b = m >> 11;
  int h = c >> 5, e = c & 31;
  ctx[n] = vm[((b * H_ + h) << 5) + e];
}

// full softmax attention for one selected query row; writes into ctx(B,L,C)
__global__ __launch_bounds__(256)
void attn_top_kernel(const float* __restrict__ q, const float* __restrict__ k,
                     const float* __restrict__ v, const int* __restrict__ top,
                     float* __restrict__ ctx)
{
  __shared__ float qs[E_];
  __shared__ float sc[L_];
  __shared__ float red[256];
  __shared__ float acc8[8][E_];
  int blk = blockIdx.x;
  int u  = blk % NTOP_;
  int bh = blk / NTOP_;
  int h  = bh % H_, b = bh / H_;
  int tid = threadIdx.x, lane = tid & 31, wave = tid >> 5;
  int l0 = top[bh * NTOP_ + u];

  const float* qp = q + ((size_t)bh * L_ + l0) * E_;
  if (tid < E_) qs[tid] = qp[tid];
  __syncthreads();

  // pass 1: scores + max
  float lmax = -3.4e38f;
  for (int j = tid; j < L_; j += 256) {
    const float* kp = k + ((size_t)bh * L_ + j) * E_;
    float s = 0.f;
#pragma unroll
    for (int e = 0; e < E_; ++e) s += qs[e] * kp[e];
    s *= SCALE_;
    sc[j] = s;
    lmax = fmaxf(lmax, s);
  }
  red[tid] = lmax;
  __syncthreads();
  for (int s = 128; s > 0; s >>= 1) {
    if (tid < s) red[tid] = fmaxf(red[tid], red[tid + s]);
    __syncthreads();
  }
  float mx = red[0];
  __syncthreads();

  // pass 2: exp + sum
  float lsum = 0.f;
  for (int j = tid; j < L_; j += 256) {
    float e_ = __expf(sc[j] - mx);
    sc[j] = e_;
    lsum += e_;
  }
  red[tid] = lsum;
  __syncthreads();
  for (int s = 128; s > 0; s >>= 1) {
    if (tid < s) red[tid] += red[tid + s];
    __syncthreads();
  }
  float denom = red[0];
  __syncthreads();

  // pass 3: weighted sum of V
  float accv[E_];
#pragma unroll
  for (int e = 0; e < E_; ++e) accv[e] = 0.f;
  for (int j = tid; j < L_; j += 256) {
    float p = sc[j];
    const float* vp = v + ((size_t)bh * L_ + j) * E_;
#pragma unroll
    for (int e = 0; e < E_; ++e) accv[e] += p * vp[e];
  }
  // wave-level allreduce per element; lane e keeps element e
  float stash = 0.f;
#pragma unroll
  for (int e = 0; e < E_; ++e) {
    float s = accv[e];
#pragma unroll
    for (int m = 16; m > 0; m >>= 1) s += __shfl_xor(s, m, 32);
    if (lane == e) stash = s;
  }
  acc8[wave][lane] = stash;
  __syncthreads();
  if (tid < E_) {
    float s = 0.f;
    for (int w = 0; w < 8; ++w) s += acc8[w][tid];
    ctx[((size_t)(b * L_ + l0)) * C_ + h * E_ + tid] = s / denom;
  }
}

// ---------------------------------------------------------------------------
// LayerNorm (row of 256), fused FFN+residual+LN2, batchnorm+elu
// ---------------------------------------------------------------------------
__global__ __launch_bounds__(256)
void ln_kernel(const float* __restrict__ in, const float* __restrict__ g,
               const float* __restrict__ b, float* __restrict__ out)
{
  __shared__ float red[256];
  int row = blockIdx.x, tid = threadIdx.x;
  float x = in[(size_t)row * C_ + tid];
  red[tid] = x;
  __syncthreads();
  for (int s = 128; s > 0; s >>= 1) { if (tid < s) red[tid] += red[tid + s]; __syncthreads(); }
  float mean = red[0] * (1.0f / C_);
  __syncthreads();
  float d = x - mean;
  red[tid] = d * d;
  __syncthreads();
  for (int s = 128; s > 0; s >>= 1) { if (tid < s) red[tid] += red[tid + s]; __syncthreads(); }
  float var = red[0] * (1.0f / C_);
  out[(size_t)row * C_ + tid] = d * rsqrtf(var + 1e-5f) * g[tid] + b[tid];
}

__global__ __launch_bounds__(256)
void ffn_ln_kernel(const float* __restrict__ xin,
                   const float* __restrict__ w1, const float* __restrict__ b1,
                   const float* __restrict__ w2, const float* __restrict__ b2,
                   const float* __restrict__ g,  const float* __restrict__ bb,
                   float* __restrict__ out)
{
  __shared__ float xs[C_];
  __shared__ float hs[INTER_];
  __shared__ float red[256];
  int row = blockIdx.x, tid = threadIdx.x;
  xs[tid] = xin[(size_t)row * C_ + tid];
  __syncthreads();
  if (tid < INTER_) {
    float a = b1[tid];
    for (int c = 0; c < C_; ++c) a += xs[c] * w1[c * INTER_ + tid];
    hs[tid] = fmaxf(a, 0.f);
  }
  __syncthreads();
  float y = b2[tid];
#pragma unroll
  for (int kk = 0; kk < INTER_; ++kk) y += hs[kk] * w2[kk * C_ + tid];
  y += xs[tid];
  red[tid] = y;
  __syncthreads();
  for (int s = 128; s > 0; s >>= 1) { if (tid < s) red[tid] += red[tid + s]; __syncthreads(); }
  float mean = red[0] * (1.0f / C_);
  __syncthreads();
  float d = y - mean;
  red[tid] = d * d;
  __syncthreads();
  for (int s = 128; s > 0; s >>= 1) { if (tid < s) red[tid] += red[tid + s]; __syncthreads(); }
  float var = red[0] * (1.0f / C_);
  out[(size_t)row * C_ + tid] = d * rsqrtf(var + 1e-5f) * g[tid] + bb[tid];
}

__global__ __launch_bounds__(256)
void bn_stats_kernel(const float* __restrict__ x, float* __restrict__ mean,
                     float* __restrict__ var)
{
  __shared__ float rs[256], rq[256];
  int o = blockIdx.x, tid = threadIdx.x;
  float s = 0.f, q = 0.f;
  for (int m = tid; m < MROWS; m += 256) {
    float v = x[(size_t)m * C_ + o];
    s += v; q += v * v;
  }
  rs[tid] = s; rq[tid] = q;
  __syncthreads();
  for (int st = 128; st > 0; st >>= 1) {
    if (tid < st) { rs[tid] += rs[tid + st]; rq[tid] += rq[tid + st]; }
    __syncthreads();
  }
  if (tid == 0) {
    float mu = rs[0] * (1.0f / MROWS);
    mean[o] = mu;
    var[o]  = rq[0] * (1.0f / MROWS) - mu * mu;
  }
}

__global__ __launch_bounds__(256)
void bn_elu_kernel(const float* __restrict__ x, const float* __restrict__ mean,
                   const float* __restrict__ var, const float* __restrict__ g,
                   const float* __restrict__ b, float* __restrict__ out)
{
  int n = blockIdx.x * 256 + threadIdx.x;
  if (n >= MROWS * C_) return;
  int o = n & (C_ - 1);
  float v = (x[n] - mean[o]) * rsqrtf(var[o] + 1e-5f) * g[o] + b[o];
  out[n] = v > 0.f ? v : (__expf(v) - 1.f);
}

// ---------------------------------------------------------------------------
// Host orchestration
// ---------------------------------------------------------------------------
extern "C" void kernel_launch(void* const* d_in, const int* in_sizes, int n_in,
                              void* d_out, int out_size, void* d_ws, size_t ws_size,
                              hipStream_t stream)
{
  (void)in_sizes; (void)n_in; (void)out_size; (void)ws_size;
  // inputs: x, idx_sample, then params dict in insertion order
  const float* x_in   = (const float*)d_in[0];
  const int*   idx_in = (const int*)d_in[1];
  const float* wq = (const float*)d_in[2];  const float* bq = (const float*)d_in[3];
  const float* wk = (const float*)d_in[4];  const float* bk = (const float*)d_in[5];
  const float* wv = (const float*)d_in[6];  const float* bv = (const float*)d_in[7];
  const float* wo = (const float*)d_in[8];  const float* bo = (const float*)d_in[9];
  const float* w1 = (const float*)d_in[10]; const float* b1 = (const float*)d_in[11];
  const float* w2 = (const float*)d_in[12]; const float* b2 = (const float*)d_in[13];
  const float* ln1g = (const float*)d_in[14]; const float* ln1b = (const float*)d_in[15];
  const float* ln2g = (const float*)d_in[16]; const float* ln2b = (const float*)d_in[17];
  const float* dconv_w = (const float*)d_in[18]; const float* dconv_b = (const float*)d_in[19];
  const float* bn_g = (const float*)d_in[20]; const float* bn_b = (const float*)d_in[21];
  const float* fln_g = (const float*)d_in[22]; const float* fln_b = (const float*)d_in[23];

  char* ws = (char*)d_ws;
  size_t off = 0;
  auto alloc = [&](size_t bytes) -> char* {
    char* p = ws + off;
    off += (bytes + 255) & ~(size_t)255;
    return p;
  };
  unsigned short* XBF = (unsigned short*)alloc((size_t)MROWS * 3 * C_ * 2); // im2col max
  unsigned short* WBF = (unsigned short*)alloc((size_t)3 * C_ * C_ * 2);    // transposed weights
  float* Q   = (float*)alloc((size_t)MROWS * C_ * 4);
  float* Kb  = (float*)alloc((size_t)MROWS * C_ * 4);
  float* V   = (float*)alloc((size_t)MROWS * C_ * 4);
  float* CTX = (float*)alloc((size_t)MROWS * C_ * 4);
  float* XA  = (float*)alloc((size_t)MROWS * C_ * 4);
  float* XB  = (float*)alloc((size_t)MROWS * C_ * 4);
  float* MB  = (float*)alloc((size_t)B_ * H_ * L_ * 4);
  int*   TOP = (int*)alloc((size_t)B_ * H_ * NTOP_ * 4);
  float* VM  = (float*)alloc((size_t)B_ * H_ * E_ * 4);
  float* BNM = (float*)alloc(C_ * 4);
  float* BNV = (float*)alloc(C_ * 4);

  const int NELEM = MROWS * C_;
  const dim3 gemm_grid(C_ / TN, MROWS / TM);

  copy_f32_kernel<<<(NELEM + 255) / 256, 256, 0, stream>>>(x_in, XA, NELEM);

  for (int i = 0; i < NL_; ++i) {
    const float* wq_i = wq + (size_t)i * C_ * C_;
    const float* wk_i = wk + (size_t)i * C_ * C_;
    const float* wv_i = wv + (size_t)i * C_ * C_;
    const float* wo_i = wo + (size_t)i * C_ * C_;
    const float* bq_i = bq + (size_t)i * C_;
    const float* bk_i = bk + (size_t)i * C_;
    const float* bv_i = bv + (size_t)i * C_;
    const float* bo_i = bo + (size_t)i * C_;
    const int*   idx_i = idx_in + (size_t)i * L_ * NTOP_;

    // ---- QKV projections (WMMA bf16, f32 accumulate; weights transposed) ----
    cvt_bf16_kernel<<<(NELEM + 255) / 256, 256, 0, stream>>>(XA, XBF, NELEM);
    cvtT_bf16_kernel<<<(C_ * C_ + 255) / 256, 256, 0, stream>>>(wq_i, WBF, C_, C_);
    wmma_gemm_kernel<<<gemm_grid, 256, 0, stream>>>(XBF, WBF, bq_i, nullptr, Q,
                                                    MROWS, C_, C_, 1, 0);
    cvtT_bf16_kernel<<<(C_ * C_ + 255) / 256, 256, 0, stream>>>(wk_i, WBF, C_, C_);
    wmma_gemm_kernel<<<gemm_grid, 256, 0, stream>>>(XBF, WBF, bk_i, nullptr, Kb,
                                                    MROWS, C_, C_, 1, 0);
    cvtT_bf16_kernel<<<(C_ * C_ + 255) / 256, 256, 0, stream>>>(wv_i, WBF, C_, C_);
    wmma_gemm_kernel<<<gemm_grid, 256, 0, stream>>>(XBF, WBF, bv_i, nullptr, V,
                                                    MROWS, C_, C_, 1, 0);

    // ---- ProbSparse attention ----
    sampled_m_kernel<<<(B_ * H_ * L_ + 255) / 256, 256, 0, stream>>>(Q, Kb, idx_i, MB);
    topk_kernel<<<B_ * H_, 256, 0, stream>>>(MB, TOP);
    vmean_kernel<<<B_ * H_, 256, 0, stream>>>(V, VM);
    ctx_fill_kernel<<<(NELEM + 255) / 256, 256, 0, stream>>>(VM, CTX);
    attn_top_kernel<<<B_ * H_ * NTOP_, 256, 0, stream>>>(Q, Kb, V, TOP, CTX);

    // ---- output projection + residual, LN1 ----
    cvt_bf16_kernel<<<(NELEM + 255) / 256, 256, 0, stream>>>(CTX, XBF, NELEM);
    cvtT_bf16_kernel<<<(C_ * C_ + 255) / 256, 256, 0, stream>>>(wo_i, WBF, C_, C_);
    wmma_gemm_kernel<<<gemm_grid, 256, 0, stream>>>(XBF, WBF, bo_i, XA, XB,
                                                    MROWS, C_, C_, 0, 0);
    ln_kernel<<<MROWS, 256, 0, stream>>>(XB, ln1g + (size_t)i * C_, ln1b + (size_t)i * C_, XB);

    // ---- fused FFN + residual + LN2 ----
    ffn_ln_kernel<<<MROWS, 256, 0, stream>>>(XB,
        w1 + (size_t)i * C_ * INTER_, b1 + (size_t)i * INTER_,
        w2 + (size_t)i * INTER_ * C_, b2 + (size_t)i * C_,
        ln2g + (size_t)i * C_, ln2b + (size_t)i * C_, XA);

    // ---- distill: 3-tap conv as K=768 WMMA GEMM, then BN + ELU ----
    if (i < NL_ - 1) {
      im2col_kernel<<<(MROWS * 3 * C_ + 255) / 256, 256, 0, stream>>>(XA, XBF);
      convw_kernel<<<(3 * C_ * C_ + 255) / 256, 256, 0, stream>>>(
          dconv_w + (size_t)i * C_ * C_ * 3, WBF);
      wmma_gemm_kernel<<<gemm_grid, 256, 0, stream>>>(XBF, WBF,
          dconv_b + (size_t)i * C_, nullptr, XB, MROWS, 3 * C_, C_, 0, 0);
      bn_stats_kernel<<<C_, 256, 0, stream>>>(XB, BNM, BNV);
      bn_elu_kernel<<<(NELEM + 255) / 256, 256, 0, stream>>>(XB, BNM, BNV,
          bn_g + (size_t)i * C_, bn_b + (size_t)i * C_, XA);
    }
  }

  // ---- final LN ----
  ln_kernel<<<MROWS, 256, 0, stream>>>(XA, fln_g, fln_b, (float*)d_out);
}
